// GCNReg_add_33243046871479
// MI455X (gfx1250) — matrix operand order repaired and checked by reading, not compiled, not run on hardware
//
#include <hip/hip_runtime.h>

// ---------------- problem constants (match reference) ----------------
#define NODES   50000
#define EDGES   1600000
#define HID     128         // IN_DIM == HID == 128
#define EXTRA   32
#define NGRAPH  64
#define DCAT    (HID + EXTRA)   // 160

typedef __attribute__((ext_vector_type(16))) _Float16 v16h;
typedef __attribute__((ext_vector_type(8)))  float    v8f;

// ---------------- utility: zero a float region ----------------
__global__ void k_zero(float* __restrict__ p, size_t n) {
    size_t i = (size_t)blockIdx.x * blockDim.x + threadIdx.x;
    size_t stride = (size_t)gridDim.x * blockDim.x;
    for (; i < n; i += stride) p[i] = 0.0f;
}

// ---------------- degrees via float atomics ----------------
__global__ void k_degrees(const int* __restrict__ src, const int* __restrict__ dst,
                          float* __restrict__ deg_out, float* __restrict__ deg_in, int e_cnt) {
    int e = blockIdx.x * blockDim.x + threadIdx.x;
    if (e < e_cnt) {
        atomicAdd(&deg_out[src[e]], 1.0f);
        atomicAdd(&deg_in [dst[e]], 1.0f);
    }
}

__global__ void k_norms(const float* __restrict__ deg_out, const float* __restrict__ deg_in,
                        float* __restrict__ norm_src, float* __restrict__ norm_dst, int n) {
    int i = blockIdx.x * blockDim.x + threadIdx.x;
    if (i < n) {
        norm_src[i] = rsqrtf(fmaxf(deg_out[i], 1.0f));
        norm_dst[i] = rsqrtf(fmaxf(deg_in [i], 1.0f));
    }
}

// ---------------- WMMA GEMM: Out[M x 128] = A[M x 128] @ W[128 x 128] ----------------
// Persistent grid-stride form. 256 threads = 8 waves; wave w owns N-tile w (16 cols).
// Each wave packs its 4 B fragments (whole K=128 of its 16 columns, f16) into
// registers ONCE, then streams M-tiles: the steady-state loop is A-load (b128),
// cvt f32->f16, and 4 x v_wmma_f32_16x16x32_f16 with accumulator chaining.
// Fragment packing used here (contiguous k order, same for A and B):
//   A 16x32 f16 : lane L -> row M = L&15; hi = L>>4; elems 0-15 hold K=kb+hi*16+0..15.
//   B 32x16 f16 : lane L -> col N = L&15; lanes 0-15 K=kb+0..15, lanes 16-31 K=kb+16..31.
//   C/D 16x16 f32: elem r, lane L -> M = r + 8*(L>>4), N = L&15.
__global__ __launch_bounds__(256)
void k_gemm_wmma(const float* __restrict__ A, const float* __restrict__ W,
                 float* __restrict__ Out, int M) {
    const int wave = threadIdx.x >> 5;     // 0..7  -> N tile
    const int lane = threadIdx.x & 31;
    const int row  = lane & 15;
    const int hi   = lane >> 4;            // 0 or 1
    const int col  = (wave << 4) + (lane & 15);

    // ---- preload all 4 B fragments (register-resident weights) ----
    v16h bfrag[4];
    #pragma unroll
    for (int t = 0; t < 4; ++t) {
        const float* wp = W + (size_t)(t * 32 + hi * 16) * HID + col;
        #pragma unroll
        for (int e = 0; e < 16; ++e) {
            bfrag[t][e] = (_Float16)wp[(size_t)e * HID];
        }
    }

    const int mtiles = M >> 4;             // 3125
    for (int mtile = blockIdx.x; mtile < mtiles; mtile += gridDim.x) {
        const float* __restrict__ arow = A + ((size_t)mtile * 16 + row) * HID;

        // prefetch next tile's A row while this tile computes
        int nxt = mtile + gridDim.x;
        if (nxt < mtiles) {
            __builtin_prefetch(A + ((size_t)nxt * 16 + row) * HID, 0, 1);
        }

        v8f c = {};
        #pragma unroll
        for (int t = 0; t < 4; ++t) {
            const int kb = t * 32;
            v16h a;
            const float* a0 = arow + kb + hi * 16;
            const float* a1 = arow + kb + hi * 16 + 8;
            #pragma unroll
            for (int j = 0; j < 8; ++j) {
                a[j]     = (_Float16)a0[j];
                a[8 + j] = (_Float16)a1[j];
            }
            // 8 args: (neg_a, A, neg_b, B, c_mod, C, reuse_a, reuse_b)
            c = __builtin_amdgcn_wmma_f32_16x16x32_f16(false, a, false, bfrag[t],
                                                       (short)0, c, false, false);
        }

        float* __restrict__ obase = Out + ((size_t)mtile * 16 + 8 * hi) * HID + col;
        #pragma unroll
        for (int r = 0; r < 8; ++r) {
            obase[(size_t)r * HID] = c[r];
        }
    }
}

// ---------------- edge gather/scatter: one wave32 per edge ----------------
// agg[dst] += hw[src] * norm_src[src]  (128 floats = 32 lanes x float4)
__global__ __launch_bounds__(256)
void k_edge_scatter(const float* __restrict__ hw,
                    const int* __restrict__ src, const int* __restrict__ dst,
                    const float* __restrict__ norm_src,
                    float* __restrict__ agg, int e_cnt) {
    int edge = (int)(((size_t)blockIdx.x * blockDim.x + threadIdx.x) >> 5);
    int lane = threadIdx.x & 31;
    if (edge >= e_cnt) return;
    int s = src[edge];
    int d = dst[edge];
    float ns = norm_src[s];
    const float4* srow = (const float4*)(hw + (size_t)s * HID);
    float4 v = srow[lane];                        // coalesced 512B per wave
    float* arow = agg + (size_t)d * HID + lane * 4;
    atomicAdd(arow + 0, v.x * ns);
    atomicAdd(arow + 1, v.y * ns);
    atomicAdd(arow + 2, v.z * ns);
    atomicAdd(arow + 3, v.w * ns);
}

// ---------------- epilogue: x = relu(x * norm_dst[node] + bias[feat]) in-place ----------------
__global__ void k_relu_norm_bias(float* __restrict__ x, const float* __restrict__ norm_dst,
                                 const float* __restrict__ bias, int n) {
    size_t i = (size_t)blockIdx.x * blockDim.x + threadIdx.x;
    if (i < (size_t)n * HID) {
        int node = (int)(i >> 7);
        int f    = (int)(i & (HID - 1));
        x[i] = fmaxf(x[i] * norm_dst[node] + bias[f], 0.0f);
    }
}

// ---------------- per-graph mean pooling: one wave32 per node ----------------
__global__ __launch_bounds__(256)
void k_pool(const float* __restrict__ h2, const int* __restrict__ gid,
            float* __restrict__ sums, float* __restrict__ counts, int n) {
    int node = (int)(((size_t)blockIdx.x * blockDim.x + threadIdx.x) >> 5);
    int lane = threadIdx.x & 31;
    if (node >= n) return;
    int g = gid[node];
    const float4* row = (const float4*)(h2 + (size_t)node * HID);
    float4 v = row[lane];
    float* s = sums + (size_t)g * HID + lane * 4;
    atomicAdd(s + 0, v.x);
    atomicAdd(s + 1, v.y);
    atomicAdd(s + 2, v.z);
    atomicAdd(s + 3, v.w);
    if (lane == 0) atomicAdd(&counts[g], 1.0f);
}

// ---------------- build hg = [mean_nodes | descriptors]  (64 x 160) ----------------
__global__ void k_build_hg(const float* __restrict__ sums, const float* __restrict__ counts,
                           const float* __restrict__ desc, float* __restrict__ hg) {
    int idx = blockIdx.x * blockDim.x + threadIdx.x;
    if (idx >= NGRAPH * DCAT) return;
    int b = idx / DCAT;
    int j = idx % DCAT;
    hg[idx] = (j < HID) ? sums[b * HID + j] / fmaxf(counts[b], 1.0f)
                        : desc[b * EXTRA + (j - HID)];
}

// ---------------- tiny dense layer: out[B x J] = act(in[B x K] @ W[K x J] + bias) ----------------
__global__ void k_dense(const float* __restrict__ in, const float* __restrict__ W,
                        const float* __restrict__ bias, float* __restrict__ out,
                        int K, int J, int do_relu) {
    int b = blockIdx.x;
    int j = threadIdx.x;
    if (j >= J) return;
    const float* row = in + (size_t)b * K;
    float acc = bias[j];
    for (int k = 0; k < K; ++k) acc += row[k] * W[(size_t)k * J + j];
    if (do_relu) acc = fmaxf(acc, 0.0f);
    out[(size_t)b * J + j] = acc;
}

// ---------------- host-side orchestration ----------------
extern "C" void kernel_launch(void* const* d_in, const int* in_sizes, int n_in,
                              void* d_out, int out_size, void* d_ws, size_t ws_size,
                              hipStream_t stream) {
    (void)in_sizes; (void)n_in; (void)out_size; (void)ws_size;

    const float* h    = (const float*)d_in[0];
    const int*   src  = (const int*)  d_in[1];
    const int*   dst  = (const int*)  d_in[2];
    const int*   gid  = (const int*)  d_in[3];
    const float* desc = (const float*)d_in[4];
    const float* W1   = (const float*)d_in[5];
    const float* b1   = (const float*)d_in[6];
    const float* W2   = (const float*)d_in[7];
    const float* b2   = (const float*)d_in[8];
    const float* Wc1  = (const float*)d_in[9];
    const float* bc1  = (const float*)d_in[10];
    const float* Wc2  = (const float*)d_in[11];
    const float* bc2  = (const float*)d_in[12];
    const float* Wc3  = (const float*)d_in[13];
    const float* bc3  = (const float*)d_in[14];
    float* out = (float*)d_out;

    // workspace carve (floats): 2 big ping-pong buffers + norms + pooling + MLP temps
    float* ws       = (float*)d_ws;
    float* bufA     = ws;                                  // NODES*HID (GEMM out / hw)
    float* bufB     = bufA + (size_t)NODES * HID;          // NODES*HID (agg / layer out)
    float* deg_out  = bufB + (size_t)NODES * HID;          // NODES
    float* deg_in   = deg_out + NODES;                     // NODES
    float* norm_src = deg_in + NODES;                      // NODES
    float* norm_dst = norm_src + NODES;                    // NODES
    float* sums     = norm_dst + NODES;                    // NGRAPH*HID
    float* counts   = sums + NGRAPH * HID;                 // NGRAPH
    float* hg       = counts + NGRAPH;                     // NGRAPH*DCAT
    float* o1       = hg + NGRAPH * DCAT;                  // NGRAPH*DCAT
    float* o2       = o1 + NGRAPH * DCAT;                  // NGRAPH*DCAT

    const int TB = 256;
    const size_t NF = (size_t)NODES * HID;
    const int gemm_blocks = 640;                           // persistent; ~5 M-tiles each
    const int edge_blocks = (EDGES * 32 + TB - 1) / TB;    // wave per edge
    const int node_wave_blocks = (NODES * 32 + TB - 1) / TB;

    // degrees + norms
    k_zero<<<2048, TB, 0, stream>>>(deg_out, (size_t)2 * NODES);
    k_degrees<<<(EDGES + TB - 1) / TB, TB, 0, stream>>>(src, dst, deg_out, deg_in, EDGES);
    k_norms<<<(NODES + TB - 1) / TB, TB, 0, stream>>>(deg_out, deg_in, norm_src, norm_dst, NODES);

    // ---- layer 1 ----
    k_gemm_wmma<<<gemm_blocks, TB, 0, stream>>>(h, W1, bufA, NODES);          // hw1 = h @ W1
    k_zero<<<4096, TB, 0, stream>>>(bufB, NF);
    k_edge_scatter<<<edge_blocks, TB, 0, stream>>>(bufA, src, dst, norm_src, bufB, EDGES);
    k_relu_norm_bias<<<(int)((NF + TB - 1) / TB), TB, 0, stream>>>(bufB, norm_dst, b1, NODES); // h1

    // ---- layer 2 ----
    k_gemm_wmma<<<gemm_blocks, TB, 0, stream>>>(bufB, W2, bufA, NODES);       // hw2 = h1 @ W2
    k_zero<<<4096, TB, 0, stream>>>(bufB, NF);
    k_edge_scatter<<<edge_blocks, TB, 0, stream>>>(bufA, src, dst, norm_src, bufB, EDGES);
    k_relu_norm_bias<<<(int)((NF + TB - 1) / TB), TB, 0, stream>>>(bufB, norm_dst, b2, NODES); // h2

    // ---- pooling + concat ----
    k_zero<<<64, TB, 0, stream>>>(sums, (size_t)NGRAPH * HID + NGRAPH);
    k_pool<<<node_wave_blocks, TB, 0, stream>>>(bufB, gid, sums, counts, NODES);
    k_build_hg<<<(NGRAPH * DCAT + TB - 1) / TB, TB, 0, stream>>>(sums, counts, desc, hg);

    // ---- MLP head ----
    k_dense<<<NGRAPH, DCAT, 0, stream>>>(hg, Wc1, bc1, o1, DCAT, DCAT, 1);
    k_dense<<<NGRAPH, DCAT, 0, stream>>>(o1, Wc2, bc2, o2, DCAT, DCAT, 1);
    k_dense<<<NGRAPH, 1,    0, stream>>>(o2, Wc3, bc3, out, DCAT, 1, 0);
}
